// RecurrentGCN_644245094791
// MI455X (gfx1250) — compile-verified, matrix operand order checked by reading, not run on hardware
//
#include <hip/hip_runtime.h>

// ---------------------------------------------------------------------------
// RecurrentGCN (GConvGRU K=1) fused kernel for gfx1250 (MI455X)
//   - one wave32 owns 16 nodes for all 64 time steps (h carried in registers)
//   - gate matmuls via v_wmma_f32_16x16x32_bf16 (bf16 in, f32 accumulate)
//   - h staged in LDS as bf16: A-operand build = 2x ds_read_b128, no cvt VALU
//   - activations on the v_tanh_f32 trans pipe (bias folded into fma)
//   - x tile software-prefetched one step ahead; packed via v_cvt_pk_bf16_f32
//   - readout (leaky -> W1 -> leaky -> W2 reduction) fused per step
// ---------------------------------------------------------------------------

typedef __bf16 bf16_t;
typedef bf16_t v2bf  __attribute__((ext_vector_type(2)));
typedef bf16_t v16bf __attribute__((ext_vector_type(16)));
typedef float  v2f   __attribute__((ext_vector_type(2)));
typedef float  v8f   __attribute__((ext_vector_type(8)));

#define T_STEPS   64
#define NNODES    20000
#define FIN       8
#define HID       32
#define NEG_SLOPE 0.01f
#define BROW      40            // LDS row stride in bf16 halfwords (80B, 16B-aligned)
#define WPB       8             // waves per block
#define NTILES    (NNODES / 16) // 1250 (exact)

union BfPack { unsigned u[8]; v16bf v; };
union Bf2U   { v2bf v; unsigned u; };

// pack two f32 -> one dword of bf16 pair; prefer the packed cvt instruction
__device__ __forceinline__ unsigned pk2(float lo, float hi) {
#if __has_builtin(__builtin_amdgcn_cvt_pk_bf16_f32)
  Bf2U cv; cv.v = __builtin_amdgcn_cvt_pk_bf16_f32(lo, hi);
  return cv.u;
#else
  v2f f; f.x = lo; f.y = hi;
  Bf2U cv; cv.v = __builtin_convertvector(f, v2bf);  // matches v_cvt_pk_bf16_f32
  return cv.u;
#endif
}

__device__ __forceinline__ v8f wmma_bf16(const BfPack& a, const BfPack& b, v8f c) {
  // D(16x16 f32) = A(16x32 bf16) * B(32x16 bf16) + C
  return __builtin_amdgcn_wmma_f32_16x16x32_bf16(
      false, a.v, false, b.v, (short)0, c, false, false);
}

// Build A-operand (16x32 bf16) straight from a bf16 LDS row: two b128 reads.
// A layout (ISA 7.12.2): lane L holds row M=L%16; lanes<16 carry K {0..7,16..23},
// lanes>=16 carry K {8..15,24..31}; dword = (K even low, K odd high) = LDS order.
__device__ __forceinline__ BfPack buildA_lds(const bf16_t* row, int lane) {
  BfPack A;
  const int k0 = (lane < 16) ? 0 : 8;
  uint4 a = *(const uint4*)(row + k0);        // K = k0..k0+7
  uint4 b = *(const uint4*)(row + 16 + k0);   // K = 16+k0..16+k0+7
  A.u[0] = a.x; A.u[1] = a.y; A.u[2] = a.z; A.u[3] = a.w;
  A.u[4] = b.x; A.u[5] = b.y; A.u[6] = b.z; A.u[7] = b.w;
  return A;
}

// Build B-operand (32x16 bf16) register tile from W[kRows][32] row-major f32,
// columns colBase..colBase+15, zero-padded in K beyond kRows (done once).
__device__ __forceinline__ BfPack loadB(const float* __restrict__ W,
                                        int kRows, int colBase, int lane) {
  BfPack B;
  const int n  = (lane & 15) + colBase;
  const int k0 = (lane < 16) ? 0 : 8;
#pragma unroll
  for (int i = 0; i < 4; ++i) {
    int ka = k0 + 2 * i, kb = ka + 1;
    float a = (ka < kRows) ? W[ka * HID + n] : 0.f;
    float b = (kb < kRows) ? W[kb * HID + n] : 0.f;
    B.u[i] = pk2(a, b);
    int kc = 16 + k0 + 2 * i, kd = kc + 1;
    float c = (kc < kRows) ? W[kc * HID + n] : 0.f;
    float d = (kd < kRows) ? W[kd * HID + n] : 0.f;
    B.u[4 + i] = pk2(c, d);
  }
  return B;
}

// ---- activations: prefer the gfx1250 v_tanh_f32 trans op ----
__device__ __forceinline__ float tanh_fast(float x) {
#if __has_builtin(__builtin_amdgcn_tanhf)
  return __builtin_amdgcn_tanhf(x);
#else
  float xc = fminf(fmaxf(x, -15.f), 15.f);
  float e  = __expf(2.f * xc);
  return (e - 1.f) * __builtin_amdgcn_rcpf(e + 1.f);
#endif
}
// sigmoid(v + b): bHalf = 0.5*b, bFull = b (fallback path)
__device__ __forceinline__ float sigmoid_b(float v, float bHalf, float bFull) {
#if __has_builtin(__builtin_amdgcn_tanhf)
  return fmaf(0.5f, __builtin_amdgcn_tanhf(fmaf(0.5f, v, bHalf)), 0.5f);
#else
  (void)bHalf;
  return __builtin_amdgcn_rcpf(1.f + __expf(-(v + bFull)));
#endif
}
__device__ __forceinline__ float leakyf_(float x) {
  return fmaxf(x, NEG_SLOPE * x);   // valid leaky-relu for slope in (0,1)
}

__global__ void init_out_kernel(float* __restrict__ out,
                                const float* __restrict__ b2) {
  int i = threadIdx.x;
  if (i < T_STEPS) out[i] = b2[0];   // atomics accumulate on top of bias b2
}

__global__ __launch_bounds__(WPB * 32)
void rgcn_gru_kernel(const float* __restrict__ x,   // [T][N][F]
                     const float* __restrict__ h0,  // [N][H]
                     const float* __restrict__ Wxz, const float* __restrict__ bxz,
                     const float* __restrict__ Whz, const float* __restrict__ bhz,
                     const float* __restrict__ Wxr, const float* __restrict__ bxr,
                     const float* __restrict__ Whr, const float* __restrict__ bhr,
                     const float* __restrict__ Wxh, const float* __restrict__ bxh,
                     const float* __restrict__ Whh, const float* __restrict__ bhh,
                     const float* __restrict__ W1,  const float* __restrict__ b1,
                     const float* __restrict__ W2,
                     float* __restrict__ out)       // [64] series | [N*H] h_fin
{
  __shared__ __align__(16) bf16_t hbuf[WPB][16 * BROW];  // per-wave h staging (bf16)
  __shared__ float tpart[WPB][T_STEPS];                  // per-wave per-t partials

  const int tid      = threadIdx.x;
  const int lane     = tid & 31;
  const int wv       = tid >> 5;
  const int tile     = blockIdx.x * WPB + wv;
  const bool active  = (tile < NTILES);
  const int nodeBase = tile * 16;

  for (int i = lane; i < T_STEPS; i += 32) tpart[wv][i] = 0.f;  // inactive waves too
  __syncthreads();

  if (active) {
    const int hi = lane >> 4;     // 0: rows 0..7, 1: rows 8..15 (C-layout half)
    const int n  = lane & 15;     // C-layout column / A-layout row
    const bf16_t* myRow = &hbuf[wv][n * BROW];
    bf16_t* rowLo = &hbuf[wv][(8 * hi) * BROW];  // rows this lane publishes to

    // ---- weights -> bf16 B-operand register tiles (done once) ----
    BfPack Bxz0 = loadB(Wxz, FIN, 0, lane), Bxz1 = loadB(Wxz, FIN, 16, lane);
    BfPack Bxr0 = loadB(Wxr, FIN, 0, lane), Bxr1 = loadB(Wxr, FIN, 16, lane);
    BfPack Bxh0 = loadB(Wxh, FIN, 0, lane), Bxh1 = loadB(Wxh, FIN, 16, lane);
    BfPack Bhz0 = loadB(Whz, HID, 0, lane), Bhz1 = loadB(Whz, HID, 16, lane);
    BfPack Bhr0 = loadB(Whr, HID, 0, lane), Bhr1 = loadB(Whr, HID, 16, lane);
    BfPack Bhh0 = loadB(Whh, HID, 0, lane), Bhh1 = loadB(Whh, HID, 16, lane);

    // combined gate biases (C-layout: lane = column); half-biases for tanh-fma fold
    const float bz0 = bxz[n] + bhz[n], bz1 = bxz[16 + n] + bhz[16 + n];
    const float br0 = bxr[n] + bhr[n], br1 = bxr[16 + n] + bhr[16 + n];
    const float bg0 = bxh[n] + bhh[n], bg1 = bxh[16 + n] + bhh[16 + n];
    const float hz0 = 0.5f * bz0, hz1 = 0.5f * bz1;
    const float hr0 = 0.5f * br0, hr1 = 0.5f * br1;

    const float w1a = W1[n], w1b = W1[16 + n];
    const float b1s = b1[0];
    float w2r[8];
#pragma unroll
    for (int r = 0; r < 8; ++r) w2r[r] = W2[nodeBase + r + 8 * hi];

    // ---- initial hidden state: f32 registers (C-layout) + bf16 LDS rows ----
    v8f hc0, hc1;
#pragma unroll
    for (int r = 0; r < 8; ++r) {
      int m = r + 8 * hi;
      float a0 = h0[(size_t)(nodeBase + m) * HID + n];
      float a1 = h0[(size_t)(nodeBase + m) * HID + 16 + n];
      hc0[r] = a0; hc1[r] = a1;
      hbuf[wv][m * BROW + n]      = (bf16_t)a0;
      hbuf[wv][m * BROW + 16 + n] = (bf16_t)a1;
    }

    // ---- x A-operand: zero halves are loop-invariant; tile prefetched 1 step ----
    BfPack Ax;
#pragma unroll
    for (int i = 0; i < 8; ++i) Ax.u[i] = 0u;   // lanes>=16 and K>=8 stay zero

    float4 xa = {}, xb = {};
    if (lane < 16) {
      const float4* p = (const float4*)(x + (size_t)(nodeBase + lane) * FIN);
      xa = p[0]; xb = p[1];                      // t = 0 tile
    }

    for (int t = 0; t < T_STEPS; ++t) {
      // A_h from LDS (current hidden state); same-wave DS ordering, no barrier
      BfPack Ah = buildA_lds(myRow, lane);

      if (lane < 16) {
        // pack the current x tile (K=0..7; rest of Ax already zero)
        Ax.u[0] = pk2(xa.x, xa.y); Ax.u[1] = pk2(xa.z, xa.w);
        Ax.u[2] = pk2(xb.x, xb.y); Ax.u[3] = pk2(xb.z, xb.w);
        // prefetch next step's tile: global latency overlaps the whole GRU step
        int tn = (t + 1 < T_STEPS) ? (t + 1) : t;
        const float4* p =
            (const float4*)(x + ((size_t)tn * NNODES + nodeBase + lane) * FIN);
        xa = p[0]; xb = p[1];
      }

      // gate GEMMs (zero C inline; bias folded into activations):
      // 12 x v_wmma_f32_16x16x32_bf16 per wave per step
      v8f zero = {};
      v8f z0 = wmma_bf16(Ax, Bxz0, zero);  z0 = wmma_bf16(Ah, Bhz0, z0);
      v8f z1 = wmma_bf16(Ax, Bxz1, zero);  z1 = wmma_bf16(Ah, Bhz1, z1);
      v8f r0 = wmma_bf16(Ax, Bxr0, zero);  r0 = wmma_bf16(Ah, Bhr0, r0);
      v8f r1 = wmma_bf16(Ax, Bxr1, zero);  r1 = wmma_bf16(Ah, Bhr1, r1);
      v8f g0 = wmma_bf16(Ax, Bxh0, zero);
      v8f g1 = wmma_bf16(Ax, Bxh1, zero);

#pragma unroll
      for (int r = 0; r < 8; ++r) {
        z0[r] = sigmoid_b(z0[r], hz0, bz0); z1[r] = sigmoid_b(z1[r], hz1, bz1);
        r0[r] = sigmoid_b(r0[r], hr0, br0); r1[r] = sigmoid_b(r1[r], hr1, br1);
      }

      // p = hc .* r  -> bf16 LDS -> A-layout
#pragma unroll
      for (int r = 0; r < 8; ++r) {
        rowLo[r * BROW + n]      = (bf16_t)(hc0[r] * r0[r]);
        rowLo[r * BROW + 16 + n] = (bf16_t)(hc1[r] * r1[r]);
      }
      BfPack Ap = buildA_lds(myRow, lane);

      g0 = wmma_bf16(Ap, Bhh0, g0);
      g1 = wmma_bf16(Ap, Bhh1, g1);

      // h update + fused readout partials
      v8f s1v;
#pragma unroll
      for (int r = 0; r < 8; ++r) {
        float ht0 = tanh_fast(g0[r] + bg0);
        float ht1 = tanh_fast(g1[r] + bg1);
        float hn0 = fmaf(z0[r], hc0[r] - ht0, ht0);   // z*hc + (1-z)*ht
        float hn1 = fmaf(z1[r], hc1[r] - ht1, ht1);
        hc0[r] = hn0; hc1[r] = hn1;
        s1v[r] = fmaf(leakyf_(hn1), w1b, leakyf_(hn0) * w1a);  // W1-dot partial
      }

      // publish new h to LDS for the next step's A operand
#pragma unroll
      for (int r = 0; r < 8; ++r) {
        rowLo[r * BROW + n]      = (bf16_t)hc0[r];
        rowLo[r * BROW + 16 + n] = (bf16_t)hc1[r];
      }

      // reduce W1-dot over the 16 feature lanes per half, finish linear1+linear2
      float psum = 0.f;
#pragma unroll
      for (int r = 0; r < 8; ++r) {
        float s = s1v[r];
        s += __shfl_xor(s, 1, 32);
        s += __shfl_xor(s, 2, 32);
        s += __shfl_xor(s, 4, 32);
        s += __shfl_xor(s, 8, 32);
        float a2 = leakyf_(s + b1s);
        psum = fmaf(a2, w2r[r], psum);
      }
      psum += __shfl_xor(psum, 16, 32);   // combine node halves
      if (lane == 0) tpart[wv][t] = psum;
    }

    // final hidden state -> out[64 + node*32 + feat]
    float* hf = out + T_STEPS;
#pragma unroll
    for (int r = 0; r < 8; ++r) {
      int m = r + 8 * hi;
      hf[(size_t)(nodeBase + m) * HID + n]      = hc0[r];
      hf[(size_t)(nodeBase + m) * HID + 16 + n] = hc1[r];
    }
  }

  __syncthreads();
  if (tid < T_STEPS) {
    float s = 0.f;
#pragma unroll
    for (int w = 0; w < WPB; ++w) s += tpart[w][tid];
    atomicAdd(out + tid, s);   // one atomic per t per block (157 blocks)
  }
}

extern "C" void kernel_launch(void* const* d_in, const int* in_sizes, int n_in,
                              void* d_out, int out_size, void* d_ws, size_t ws_size,
                              hipStream_t stream) {
  // setup_inputs() order:
  //  0 x [T,N,F] f32      1 edge_index i64 (unused, K=1)   2 edge_weight f32 (unused)
  //  3 h [N,H] f32        4 Wxz  5 bxz  6 Whz  7 bhz
  //  8 Wxr  9 bxr 10 Whr 11 bhr 12 Wxh 13 bxh 14 Whh 15 bhh
  // 16 W1 [H,1] 17 b1 [1] 18 W2 [N,1] 19 b2 [1]
  const float* x   = (const float*)d_in[0];
  const float* h0  = (const float*)d_in[3];
  const float* Wxz = (const float*)d_in[4],  *bxz = (const float*)d_in[5];
  const float* Whz = (const float*)d_in[6],  *bhz = (const float*)d_in[7];
  const float* Wxr = (const float*)d_in[8],  *bxr = (const float*)d_in[9];
  const float* Whr = (const float*)d_in[10], *bhr = (const float*)d_in[11];
  const float* Wxh = (const float*)d_in[12], *bxh = (const float*)d_in[13];
  const float* Whh = (const float*)d_in[14], *bhh = (const float*)d_in[15];
  const float* W1  = (const float*)d_in[16], *b1  = (const float*)d_in[17];
  const float* W2  = (const float*)d_in[18], *b2  = (const float*)d_in[19];
  float* out = (float*)d_out;

  init_out_kernel<<<1, 64, 0, stream>>>(out, b2);

  const int blocks = (NTILES + WPB - 1) / WPB;   // 157
  rgcn_gru_kernel<<<blocks, WPB * 32, 0, stream>>>(
      x, h0, Wxz, bxz, Whz, bhz, Wxr, bxr, Whr, bhr,
      Wxh, bxh, Whh, bhh, W1, b1, W2, out);
}
